// RNNModel_71055938945282
// MI455X (gfx1250) — matrix operand order repaired and checked by reading, not compiled, not run on hardware
//
#include <hip/hip_runtime.h>
#include <math.h>

// ---------------------------------------------------------------------------
// Leaky-integrator RNN on gfx1250 (MI455X).
//   x_{t+1} = x_t + alpha*(-x_t + u_t W_rec^T + I_t W_in^T + b), u=tanh(x)
// 16 persistent workgroups (one per WGP), each owns 64 neurons.
// W_rec slice resident in LDS as f16; u exchanged via double-buffered global
// buffer in d_ws with a generation-counter grid barrier per timestep.
// Matmul: v_wmma_f32_16x16x32_f16, f32 accumulate, x-state kept in D-fragment
// registers for the whole 1000-step recurrence.
// ---------------------------------------------------------------------------

typedef __attribute__((ext_vector_type(16))) _Float16 v16h;
typedef __attribute__((ext_vector_type(8)))  _Float16 v8h;
typedef __attribute__((ext_vector_type(8)))  float    v8f;

#define Mdim  64
#define Ndim  1024
#define Tdim  1000
#define NBLK  16
#define NSLC  64          // neurons per workgroup
#define WST   1032        // LDS W row stride in halves (16B aligned, bank-skewed)
#define UST   72          // LDS u-staging row stride in halves
#define ALPHA 0.2f        // dt/tau

__device__ __forceinline__ v16h cat8(v8h lo, v8h hi) {
  return __builtin_shufflevector(lo, hi, 0,1,2,3,4,5,6,7,8,9,10,11,12,13,14,15);
}

// --- init: zero u-buffer 0 (u_0 = tanh(0) = 0) and barrier counter ---------
__global__ void __launch_bounds__(256, 1)
rnn_init(_Float16* __restrict__ ubuf, unsigned* __restrict__ cnt) {
  unsigned i = blockIdx.x * blockDim.x + threadIdx.x;
  if (i == 0) *cnt = 0u;
  for (unsigned j = i; j < Mdim * Ndim; j += gridDim.x * blockDim.x)
    ubuf[j] = (_Float16)0.0f;
}

// --- persistent step kernel -------------------------------------------------
__global__ void __launch_bounds__(256, 1)
rnn_persistent(const float* __restrict__ I,     // (64, 2, 1000)
               const float* __restrict__ Win,   // (1024, 2)
               const float* __restrict__ Wrec,  // (1024, 1024)
               const float* __restrict__ bvec,  // (1024)
               const float* __restrict__ Wout,  // (2, 1024)
               float* __restrict__ out_x,       // (64, 1024, 1000)
               float* __restrict__ out_u,       // (64, 1024, 1000)
               float* __restrict__ out_y,       // (64, 2, 1000)
               _Float16* __restrict__ ubuf,     // 2 * 64*1024 halves (d_ws)
               unsigned* __restrict__ cnt)      // barrier generation counter
{
  __shared__ __align__(16) _Float16 sW[NSLC * WST];  // W_rec slice, f16 (~129KB)
  __shared__ __align__(16) _Float16 sU[Mdim * UST];  // u_new slice staging (9KB)
  __shared__ float sIt[Mdim * 2];                    // I[:, :, t] staging

  const int tid   = threadIdx.x;
  const int blk   = blockIdx.x;
  const int nbase = blk * NSLC;

  // Load our W_rec rows (n in slice, all k) -> f16 LDS, read HBM exactly once.
  for (int idx = tid; idx < NSLC * Ndim; idx += 256) {
    int nl = idx >> 10, k = idx & (Ndim - 1);
    sW[nl * WST + k] = (_Float16)Wrec[(size_t)(nbase + nl) * Ndim + k];
  }

  const int w     = tid >> 5;          // wave 0..7
  const int lane  = tid & 31;
  const int lhi   = lane >> 4;         // 0/1: which half of the wave
  const int lln   = lane & 15;
  const int mtile = w & 3;             // wave owns M-tile mtile,
  const int ntp   = w >> 2;            // and N-tiles {2*ntp, 2*ntp+1}
  const int nloc0 = (2 * ntp) * 16 + lln;
  const int nloc1 = (2 * ntp + 1) * 16 + lln;
  const int mbase = mtile * 16 + 8 * lhi;  // C/D fragment: row = mbase + r
  const int marow = mtile * 16 + lln;      // A fragment: this lane's u row

  // Per-lane input-drive constants (n fixed per lane per tile).
  const float winA0 = Win[(size_t)(nbase + nloc0) * 2 + 0];
  const float winB0 = Win[(size_t)(nbase + nloc0) * 2 + 1];
  const float bb0   = bvec[nbase + nloc0];
  const float winA1 = Win[(size_t)(nbase + nloc1) * 2 + 0];
  const float winB1 = Win[(size_t)(nbase + nloc1) * 2 + 1];
  const float bb1   = bvec[nbase + nloc1];

  // Readout assignment: 8 waves -> (4 batch rows) x (2 outputs) per block.
  const int ym = blk * 4 + (w >> 1);
  const int yo = w & 1;

  auto compute_y = [&](const _Float16* uc, int tw) {
    float s = 0.f;
    const _Float16* up = uc + (size_t)ym * Ndim;
    const float*    wp = Wout + (size_t)yo * Ndim;
#pragma unroll 8
    for (int k0 = 0; k0 < Ndim; k0 += 32)
      s += (float)up[k0 + lane] * wp[k0 + lane];
#pragma unroll
    for (int off = 16; off > 0; off >>= 1) s += __shfl_xor(s, off, 32);
    if (lane == 0) out_y[((size_t)ym * 2 + yo) * Tdim + tw] = s;
  };

  // State held in WMMA D-fragment layout registers for the whole recurrence.
  v8f xs[2] = {};   // x_t
  v8f us[2] = {};   // u_t = tanh(x_t)  (tanh(0)=0)

  for (int t = 0; t < Tdim; ++t) {
    const _Float16* ucur = ubuf + (size_t)(t & 1) * (Mdim * Ndim);
    _Float16*       unxt = ubuf + (size_t)((t + 1) & 1) * (Mdim * Ndim);

    // Stage I[:, :, t]; prefetch next column. I is (M,2,T): elem = tid*T + t.
    if (tid < 128) {
      sIt[tid] = I[(size_t)tid * Tdim + t];
      if (t + 1 < Tdim) __builtin_prefetch(&I[(size_t)tid * Tdim + t + 1], 0, 1);
    }
    // y_{t-1} = W_out @ u_t (full u_t visible after previous barrier).
    if (t > 0) compute_y(ucur, t - 1);
    __syncthreads();

    // Emit pre-update states x_t, u_t for our (m,n) tile elements.
#pragma unroll
    for (int tt = 0; tt < 2; ++tt) {
      const int nl = tt ? nloc1 : nloc0;
#pragma unroll
      for (int r = 0; r < 8; ++r) {
        size_t off = ((size_t)(mbase + r) * Ndim + (size_t)(nbase + nl)) * Tdim + t;
        out_x[off] = xs[tt][r];
        out_u[off] = us[tt][r];
      }
    }

    // mat = u_t @ Wrec_slice^T : 32 K-steps of v_wmma_f32_16x16x32_f16.
    v8f acc0 = {}; v8f acc1 = {};
    const _Float16* arow  = ucur + (size_t)marow * Ndim + lhi * 8;
    const _Float16* brow0 = sW + (size_t)nloc0 * WST + lhi * 8;
    const _Float16* brow1 = sW + (size_t)nloc1 * WST + lhi * 8;
#pragma unroll 4
    for (int kk = 0; kk < 32; ++kk) {
      const _Float16* ap  = arow + kk * 32;           // A: lanes<16 K{0..7,16..23}, lanes>=16 K{8..15,24..31}
      v16h a = cat8(*(const v8h*)ap, *(const v8h*)(ap + 16));
      const _Float16* bp0 = brow0 + kk * 32;          // B: lanes<16 K{0..7,16..23}, lanes>=16 K{8..15,24..31}
      v16h b0 = cat8(*(const v8h*)bp0, *(const v8h*)(bp0 + 16));
      const _Float16* bp1 = brow1 + kk * 32;
      v16h b1 = cat8(*(const v8h*)bp1, *(const v8h*)(bp1 + 16));
      acc0 = __builtin_amdgcn_wmma_f32_16x16x32_f16(false, a, false, b0,
                                                    (short)0, acc0, false, false);
      acc1 = __builtin_amdgcn_wmma_f32_16x16x32_f16(false, a, false, b1,
                                                    (short)0, acc1, false, false);
    }

    // Elementwise leaky update + tanh on fragment registers.
#pragma unroll
    for (int r = 0; r < 8; ++r) {
      const float i0 = sIt[(mbase + r) * 2 + 0];
      const float i1 = sIt[(mbase + r) * 2 + 1];
      {
        float inp = fmaf(i0, winA0, fmaf(i1, winB0, bb0));
        float xo  = xs[0][r];
        float xn  = xo + ALPHA * (acc0[r] + inp - xo);
        xs[0][r] = xn;
        us[0][r] = tanhf(xn);
      }
      {
        float inp = fmaf(i0, winA1, fmaf(i1, winB1, bb1));
        float xo  = xs[1][r];
        float xn  = xo + ALPHA * (acc1[r] + inp - xo);
        xs[1][r] = xn;
        us[1][r] = tanhf(xn);
      }
    }

    // Stage u_{t+1} slice as f16 in LDS, then coalesced flush to ubuf[nxt].
#pragma unroll
    for (int r = 0; r < 8; ++r) {
      sU[(mbase + r) * UST + nloc0] = (_Float16)us[0][r];
      sU[(mbase + r) * UST + nloc1] = (_Float16)us[1][r];
    }
    __syncthreads();
    {
      const int mrow = tid >> 2, q = tid & 3;   // 4 threads x 32B per batch row
      const v8h* src = (const v8h*)&sU[mrow * UST + q * 16];
      v8h d0 = src[0], d1 = src[1];
      v8h* dst = (v8h*)(unxt + (size_t)mrow * Ndim + (size_t)(nbase + q * 16));
      dst[0] = d0; dst[1] = d1;
    }

    // Grid barrier (generation counting; counter reset by init kernel).
    __threadfence();
    __syncthreads();
    if (tid == 0) {
      __hip_atomic_fetch_add(cnt, 1u, __ATOMIC_RELEASE, __HIP_MEMORY_SCOPE_AGENT);
      const unsigned target = (unsigned)(t + 1) * NBLK;
      while (__hip_atomic_load(cnt, __ATOMIC_ACQUIRE, __HIP_MEMORY_SCOPE_AGENT) < target)
        __builtin_amdgcn_s_sleep(2);
    }
    __syncthreads();
    __threadfence();
  }

  // Final readout: y_{T-1} from u_T (in buffer T%2, visible after last barrier).
  compute_y(ubuf + (size_t)(Tdim & 1) * (Mdim * Ndim), Tdim - 1);
}

// ---------------------------------------------------------------------------
extern "C" void kernel_launch(void* const* d_in, const int* in_sizes, int n_in,
                              void* d_out, int out_size, void* d_ws, size_t ws_size,
                              hipStream_t stream) {
  (void)in_sizes; (void)n_in; (void)out_size; (void)ws_size;
  const float* I    = (const float*)d_in[0];
  const float* Win  = (const float*)d_in[1];
  const float* Wrec = (const float*)d_in[2];
  const float* bvec = (const float*)d_in[3];
  const float* Wout = (const float*)d_in[4];

  float* out_x = (float*)d_out;                       // 64*1024*1000
  float* out_u = out_x + (size_t)Mdim * Ndim * Tdim;  // 64*1024*1000
  float* out_y = out_u + (size_t)Mdim * Ndim * Tdim;  // 64*2*1000

  _Float16* ubuf = (_Float16*)d_ws;                         // 2 * 128KB
  unsigned* cnt  = (unsigned*)((char*)d_ws + 2 * (size_t)Mdim * Ndim * sizeof(_Float16));

  rnn_init<<<64, 256, 0, stream>>>(ubuf, cnt);
  rnn_persistent<<<NBLK, 256, 0, stream>>>(I, Win, Wrec, bvec, Wout,
                                           out_x, out_u, out_y, ubuf, cnt);
}
